// ConditionalRBM_51642686767523
// MI455X (gfx1250) — matrix operand order repaired and checked by reading, not compiled, not run on hardware
//
#include <hip/hip_runtime.h>
#include <hip/hip_fp16.h>

typedef __attribute__((ext_vector_type(16))) _Float16     v16h;
typedef __attribute__((ext_vector_type(8)))  float        v8f;
typedef __attribute__((ext_vector_type(4)))  unsigned int v4u;
typedef __attribute__((ext_vector_type(8)))  int          v8i;
typedef __attribute__((ext_vector_type(4)))  int          v4i;

#define B_TOT   8192
#define NU      64
#define NV      256
#define NH      256
#define THERM   24
#define NSAMP   8
#define M_TILE  64          // batch rows per workgroup (4 waves x 16 rows)
#define THREADS 128

// ---- LDS layout (bytes) ----
// Wt_pack: ph direction  B[k=v][n=h] = Whv[h][v]   16 ntiles x 8 ktiles x 1KB
// W_pack : pv direction  B[k=h][n=v] = Whv[h][v]   16 ntiles x 8 ktiles x 1KB
// stage  : per-wave 16x256 state in A-fragment layout, 8KB/wave
// chunk  : 2 x 16KB TDM double buffer (16 rows of f32 W_hv each)
#define WT_PACK_OFF  0
#define W_PACK_OFF   131072
#define STAGE_OFF    262144
#define CHUNK_OFF    294912
#define LDS_BYTES    327680      // full 320 KB WGP LDS

// ---- d_ws layout (f16) ----
#define WHU_OFF_H    0           // 16 ntiles x 2 ktiles x 512 f16 (32 KB)
#define WVU_OFF_H    16384       // same size

__device__ __forceinline__ v8f wmma_f16(v16h a, v16h b, v8f c) {
  // 8 args: (neg_a, A, neg_b, B, c_mod, C, reuse_a, reuse_b)
  return __builtin_amdgcn_wmma_f32_16x16x32_f16(false, a, false, b, (short)0, c, false, false);
}

__device__ __forceinline__ float urand(unsigned a, unsigned b) {
  unsigned x = a * 0x9E3779B9u + b * 0x85EBCA77u;
  x ^= x >> 16; x *= 0x7FEB352Du;
  x ^= x >> 15; x *= 0x846CA68Bu;
  x ^= x >> 16;
  return (float)(x >> 8) * (1.0f / 16777216.0f);
}

// TDM: async-load one 16-row chunk of W_hv (f32, 16x256) into LDS at bufOff.
// D# bitfields per CDNA5 ISA 8.3/8.4 (2D tensor, data_size=4B, stride 256).
__device__ __forceinline__ void tdm_load_chunk(const float* Whv, int c, unsigned bufOff) {
  unsigned long long ga = (unsigned long long)(const void*)(Whv + (size_t)c * 16 * NV);
  v4u g0 = { 1u,                                   // count=1, user descriptor
             bufOff,                               // lds_addr (bytes)
             (unsigned)(ga & 0xffffffffu),         // global_addr[31:0]
             (unsigned)((ga >> 32) & 0x01ffffffu)  // global_addr[56:32]
               | (2u << 30) };                     // type=2 ("image")
  v8i g1 = { (int)(2u << 16),   // workgroup_mask=0, data_size=2 (4 bytes)
             0x01000000,        // tensor_dim0=256 (low16 in bits 63:48)
             0x00100000,        // tensor_dim0 hi=0 | tensor_dim1=16 (bits 95:80)
             (int)(256u << 16), // tensor_dim1 hi=0 | tile_dim0=256
             16,                // tile_dim1=16, tile_dim2=0
             256,               // tensor_dim0_stride=256 (lo 32)
             0, 0 };            // stride hi / tensor_dim1_stride = 0
  v4i z4 = {0, 0, 0, 0};
#if __clang_major__ >= 23
  v8i z8 = {0, 0, 0, 0, 0, 0, 0, 0};
  __builtin_amdgcn_tensor_load_to_lds(g0, g1, z4, z4, z8, 0);
#else
  __builtin_amdgcn_tensor_load_to_lds(g0, g1, z4, z4, 0);
#endif
}

// ------------------------------------------------------------------
// Prep: pack W_hu (ph) and W_vu (pv) into B-fragment-native f16 layout
// ------------------------------------------------------------------
__global__ void rbm_prep_kernel(const float* __restrict__ Whu,
                                const float* __restrict__ Wvu,
                                _Float16* __restrict__ ws) {
  int i = blockIdx.x * blockDim.x + threadIdx.x;   // 0 .. 32767
  int pack  = i >> 14;            // 0 = W_hu pack, 1 = W_vu pack
  int rem   = i & 16383;
  int block = rem >> 9;           // 0..31  (nt*2 + kt)
  int elem  = rem & 511;
  int nt = block >> 1, kt = block & 1;
  int lane_b = elem >> 4, slot = elem & 15;
  int n = lane_b & 15;
  int k = slot + ((lane_b >> 4) << 4);
  int ng = nt * 16 + n;           // output index (h or v)
  int kg = kt * 32 + k;           // u index
  float w = (pack == 0) ? Whu[ng * NU + kg] : Wvu[ng * NU + kg];
  ws[i] = (_Float16)w;            // linear index == packed layout
}

// ------------------------------------------------------------------
// Main persistent chain kernel: one workgroup owns 64 batch rows for
// the entire 32-sweep Gibbs chain.
// ------------------------------------------------------------------
__global__ void __launch_bounds__(THREADS)
rbm_chain_kernel(const float* __restrict__ u_state,
                 const float* __restrict__ bias_v,
                 const float* __restrict__ bias_h,
                 const float* __restrict__ Whv,
                 const _Float16* __restrict__ ws,
                 float* __restrict__ out) {
  extern __shared__ char smem[];
  _Float16* lds = (_Float16*)smem;

  const int tid   = threadIdx.x;
  const int lane  = tid & 31;
  const int wave  = tid >> 5;
  const int hi    = lane >> 4;        // lane half (A/D layout selector)
  const int mlane = lane & 15;
  const int rowBase = blockIdx.x * M_TILE + wave * 16;

  // ---- Phase 0: TDM-stream W_hv into LDS, repack into both orientations ----
  if (wave == 0) tdm_load_chunk(Whv, 0, CHUNK_OFF);
#pragma unroll 1
  for (int c = 0; c < 16; ++c) {
    if (wave == 0) {
      if (c < 15) {
        tdm_load_chunk(Whv, c + 1, CHUNK_OFF + (unsigned)(((c + 1) & 1) * 16384));
        __builtin_amdgcn_s_wait_tensorcnt(1);   // chunk c landed
      } else {
        __builtin_amdgcn_s_wait_tensorcnt(0);
      }
    }
    __syncthreads();                            // chunk c visible to all waves
    const float* cb = (const float*)(smem + CHUNK_OFF + (c & 1) * 16384);
    // pack0 (ph, Wt): B[k=v][n=h]; this chunk supplies nt == c (h rows)
    for (int i = tid; i < 4096; i += THREADS) {
      int kt = i >> 9, elem = i & 511;
      int lane_b = elem >> 4, slot = elem & 15;
      int n = lane_b & 15;
      int k = slot + ((lane_b >> 4) << 4);
      lds[(c * 8 + kt) * 512 + elem] = (_Float16)cb[n * NV + kt * 32 + k];
    }
    // pack1 (pv, W): B[k=h][n=v]; this chunk supplies k-half (c>>1, c&1)
    for (int i = tid; i < 4096; i += THREADS) {
      int nt = i >> 8, e2 = i & 255;
      int lane_low = e2 >> 4, slot = e2 & 15;
      int elem = (lane_low + ((c & 1) << 4)) * 16 + slot;
      lds[W_PACK_OFF / 2 + (nt * 8 + (c >> 1)) * 512 + elem] =
          (_Float16)cb[slot * NV + nt * 16 + lane_low];
    }
    __syncthreads();                            // buf free before TDM reuses it
  }

  const _Float16* wtp   = lds;                      // ph W pack
  const _Float16* wp    = lds + (W_PACK_OFF / 2);   // pv W pack
  _Float16*       stage = lds + (STAGE_OFF / 2) + wave * 4096;
  const _Float16* whu   = ws + WHU_OFF_H;
  const _Float16* wvu   = ws + WVU_OFF_H;

  // ---- u A-fragments (K=64 -> 2 fragments), resident for whole kernel ----
  v16h ufrag[2];
#pragma unroll
  for (int f = 0; f < 2; ++f) {
    v16h t;
#pragma unroll
    for (int s = 0; s < 16; ++s) {
      int k = s + (hi << 4) + f * 32;
      t[s] = (_Float16)u_state[(rowBase + mlane) * NU + k];
    }
    ufrag[f] = t;
  }

  // sample from D-layout acc, write sample into A-fragment-layout staging.
  // rnd < 1/(1+e)  <=>  rnd*(1+e) < 1   (division-free sigmoid sampling)
  auto sampleStore = [&](v8f acc, int nt, unsigned seed, float* outp) {
#pragma unroll
    for (int r = 0; r < 8; ++r) {
      int md  = r + hi * 8;                  // row within wave tile (D layout)
      int col = nt * 16 + mlane;             // output column (becomes K next)
      float e = __expf(-acc[r]);
      unsigned grow = (unsigned)(rowBase + md);
      float rnd = urand(seed, grow * 1024u + (unsigned)col);
      float s = (__builtin_fmaf(rnd, e, rnd) < 1.0f) ? 1.0f : 0.0f;
      // D -> A index transform
      int kk     = col & 31;
      int kt     = col >> 5;
      int lane_a = md + ((kk >> 4) << 4);
      int slot   = kk & 15;
      stage[kt * 512 + lane_a * 16 + slot] = (_Float16)s;
      if (outp) outp[grow * NV + col] = s;
    }
  };

  // ---- init: v0 ~ Bernoulli(sigmoid(u @ Wvu^T + b_v)) ----
  for (int nt = 0; nt < 16; ++nt) {
    float b = bias_v[nt * 16 + mlane];
    v8f acc = {b, b, b, b, b, b, b, b};
#pragma unroll
    for (int f = 0; f < 2; ++f) {
      v16h bw = *(const v16h*)(wvu + (nt * 2 + f) * 512 + lane * 16);
      acc = wmma_f16(ufrag[f], bw, acc);
    }
    sampleStore(acc, nt, 1u, nullptr);
  }

  // ---- Gibbs chain: 24 thermalization + 8 sample sweeps ----
#pragma unroll 1
  for (int step = 0; step < THERM + NSAMP; ++step) {
    // -------- half-step 1: h | v,u --------
    v16h afrag[8];
    __builtin_prefetch(whu, 0, 0);
#pragma unroll
    for (int f = 0; f < 8; ++f)
      afrag[f] = *(const v16h*)(stage + f * 512 + lane * 16);
    for (int nt = 0; nt < 16; ++nt) {
      float b = bias_h[nt * 16 + mlane];
      v8f acc = {b, b, b, b, b, b, b, b};
#pragma unroll
      for (int f = 0; f < 2; ++f) {
        v16h bw = *(const v16h*)(whu + (nt * 2 + f) * 512 + lane * 16);
        acc = wmma_f16(ufrag[f], bw, acc);
      }
#pragma unroll
      for (int kt = 0; kt < 8; ++kt) {
        v16h bw = *(const v16h*)(wtp + (nt * 8 + kt) * 512 + lane * 16);
        acc = wmma_f16(afrag[kt], bw, acc);
      }
      sampleStore(acc, nt, 2u + 2u * (unsigned)step, nullptr);
    }

    // -------- half-step 2: v | h,u --------
    __builtin_prefetch(wvu, 0, 0);
#pragma unroll
    for (int f = 0; f < 8; ++f)
      afrag[f] = *(const v16h*)(stage + f * 512 + lane * 16);
    float* outp = (step >= THERM)
                    ? out + (size_t)(step - THERM) * (size_t)B_TOT * NV
                    : nullptr;
    for (int nt = 0; nt < 16; ++nt) {
      float b = bias_v[nt * 16 + mlane];
      v8f acc = {b, b, b, b, b, b, b, b};
#pragma unroll
      for (int f = 0; f < 2; ++f) {
        v16h bw = *(const v16h*)(wvu + (nt * 2 + f) * 512 + lane * 16);
        acc = wmma_f16(ufrag[f], bw, acc);
      }
#pragma unroll
      for (int kt = 0; kt < 8; ++kt) {
        v16h bw = *(const v16h*)(wp + (nt * 8 + kt) * 512 + lane * 16);
        acc = wmma_f16(afrag[kt], bw, acc);
      }
      sampleStore(acc, nt, 3u + 2u * (unsigned)step, outp);
    }
  }
}

extern "C" void kernel_launch(void* const* d_in, const int* in_sizes, int n_in,
                              void* d_out, int out_size, void* d_ws, size_t ws_size,
                              hipStream_t stream) {
  (void)in_sizes; (void)n_in; (void)out_size; (void)ws_size;
  const float* u_state = (const float*)d_in[0];
  const float* w_vu    = (const float*)d_in[1];   // [V, U]
  const float* w_hu    = (const float*)d_in[2];   // [H, U]
  const float* w_hv    = (const float*)d_in[3];   // [H, V]
  const float* bias_v  = (const float*)d_in[4];
  const float* bias_h  = (const float*)d_in[5];
  _Float16* ws = (_Float16*)d_ws;                 // 64 KB of packed f16 weights
  float* out = (float*)d_out;

  // Pack W_hu / W_vu into B-fragment layout in scratch (L2-resident).
  rbm_prep_kernel<<<128, 256, 0, stream>>>(w_hu, w_vu, ws);

  // Persistent Gibbs chain: 128 WGs x 128 threads, 320 KB dynamic LDS each.
  rbm_chain_kernel<<<B_TOT / M_TILE, THREADS, LDS_BYTES, stream>>>(
      u_state, bias_v, bias_h, w_hv, ws, out);
}
